// DebertaAttention_27839978012731
// MI455X (gfx1250) — compile-verified, matrix operand order checked by reading, not run on hardware
//
#include <hip/hip_runtime.h>
#include <hip/hip_bf16.h>

#define SQ 1024
#define NHEAD 16
#define DHEAD 64

typedef __attribute__((ext_vector_type(16))) __bf16 v16bf;
typedef __attribute__((ext_vector_type(8)))  float  v8f;
typedef __attribute__((ext_vector_type(4)))  int    v4i;

#if defined(__has_builtin)
#if __has_builtin(__builtin_amdgcn_global_load_async_to_lds_b128) && \
    __has_builtin(__builtin_amdgcn_s_wait_asynccnt)
#define HAVE_ASYNC_LDS 1
#endif
#endif
#ifndef HAVE_ASYNC_LDS
#define HAVE_ASYNC_LDS 0
#endif

union Frag { v16bf v; unsigned int u[8]; };
union Acc  { v8f v; float f[8]; };

__device__ __forceinline__ v8f vzero() {
  v8f z = {0.f, 0.f, 0.f, 0.f, 0.f, 0.f, 0.f, 0.f};
  return z;
}

__device__ __forceinline__ unsigned short f2bf(float x) {
  union { float f; unsigned int u; } c; c.f = x;
  unsigned int u = c.u;
  u += 0x7fffu + ((u >> 16) & 1u);      // round-to-nearest-even
  return (unsigned short)(u >> 16);
}
__device__ __forceinline__ unsigned int f2bf2(float lo, float hi) {
  return (unsigned int)f2bf(lo) | ((unsigned int)f2bf(hi) << 16);
}

// Load a 16x32 bf16 WMMA A/B fragment from a row-major 16xK tile.
// rowp = per-lane pointer to row (lane&15), already offset by kbase.
// ISA layout: VGPR g   : K = 2g + 8*half ;  VGPR 4+g : K = 16 + 2g + 8*half
__device__ __forceinline__ void load_frag(const unsigned short* rowp, int lane, Frag& f) {
  const unsigned short* p = rowp + ((lane >> 4) << 3);
#pragma unroll
  for (int g = 0; g < 4; ++g) {
    f.u[g]     = *(const unsigned int*)(p + 2 * g);
    f.u[4 + g] = *(const unsigned int*)(p + 16 + 2 * g);
  }
}

__device__ __forceinline__ void wmma_bf(v8f& acc, const Frag& a, const Frag& b) {
  acc = __builtin_amdgcn_wmma_f32_16x16x32_bf16(false, a.v, false, b.v,
                                                (short)0, acc, false, false);
}

// ---------------------------------------------------------------------------
// Generic bf16 GEMM: C[M,N] = A[M,K] @ B[N,K]^T, 64x64 tile per 128-thr block.
// K-step 64: 8 WMMAs per wave per barrier pair. bf16 A path uses async->LDS.
// mode 0: QKV epilogue   mode 1: pos-projection epilogue   mode 2: out+residual
// ---------------------------------------------------------------------------
__global__ __launch_bounds__(128) void gemm_bf16_k(
    const void* __restrict__ Ap, int a_bf16, const float* __restrict__ Bw,
    int M, int N, int K, int mode,
    const float* __restrict__ bias, const float* __restrict__ bias2,
    const float* __restrict__ resid, float scl,
    void* __restrict__ o0, void* __restrict__ o1, void* __restrict__ o2)
{
  __shared__ unsigned short As[64][72];   // stride 144B: 16B-aligned rows, no frag-read conflicts
  __shared__ unsigned short Bs[64][72];
  const int tid = threadIdx.x, lane = tid & 31, wave = tid >> 5;
  const int half = lane >> 4, nl = lane & 15;
  const int n0 = blockIdx.x * 64, m0 = blockIdx.y * 64;

  Acc acc[4];
#pragma unroll
  for (int c = 0; c < 4; ++c) acc[c].v = vzero();

  const int r = tid >> 1, c0 = (tid & 1) * 32;
  for (int kk = 0; kk < K; kk += 64) {
    __syncthreads();
    if (a_bf16) {
      const unsigned short* s = (const unsigned short*)Ap + (size_t)(m0 + r) * K + kk + c0;
#if HAVE_ASYNC_LDS
#pragma unroll
      for (int i = 0; i < 32; i += 8)
        __builtin_amdgcn_global_load_async_to_lds_b128(
            (v4i*)(s + i),
            (__attribute__((address_space(3))) v4i*)&As[r][c0 + i], 0, 0);
#else
      __builtin_prefetch(s + 64, 0, 1);
#pragma unroll
      for (int i = 0; i < 32; i += 2)
        *(unsigned int*)&As[r][c0 + i] = *(const unsigned int*)(s + i);
#endif
    } else {
      const float* s = (const float*)Ap + (size_t)(m0 + r) * K + kk + c0;
      __builtin_prefetch(s + 64, 0, 1);
#pragma unroll
      for (int i = 0; i < 32; i += 2)
        *(unsigned int*)&As[r][c0 + i] = f2bf2(s[i], s[i + 1]);
    }
    {
      const float* s = Bw + (size_t)(n0 + r) * K + kk + c0;
      __builtin_prefetch(s + 64, 0, 1);
#pragma unroll
      for (int i = 0; i < 32; i += 2)
        *(unsigned int*)&Bs[r][c0 + i] = f2bf2(s[i], s[i + 1]);
    }
#if HAVE_ASYNC_LDS
    if (a_bf16) __builtin_amdgcn_s_wait_asynccnt(0);
#endif
    __syncthreads();
    Frag a0, a1;
    load_frag(&As[wave * 16 + nl][0],  lane, a0);
    load_frag(&As[wave * 16 + nl][32], lane, a1);
#pragma unroll
    for (int ct = 0; ct < 4; ++ct) {
      Frag b0, b1;
      load_frag(&Bs[ct * 16 + nl][0],  lane, b0);
      load_frag(&Bs[ct * 16 + nl][32], lane, b1);
      wmma_bf(acc[ct].v, a0, b0);
      wmma_bf(acc[ct].v, a1, b1);
    }
  }

#pragma unroll
  for (int ct = 0; ct < 4; ++ct) {
#pragma unroll
    for (int g = 0; g < 8; ++g) {
      const int m = m0 + wave * 16 + g + 8 * half;   // C/D: VGPR g -> M = g + 8*half
      const int n = n0 + ct * 16 + nl;               // lane -> N
      const float f = acc[ct].f[g];
      if (mode == 0) {                               // QKV
        const int bb = m >> 10, s = m & 1023;
        if (n < 1024) {                              // q: (+bias)*inv_scale, [b,h,s,d]
          const int hh = n >> 6, d = n & 63;
          ((unsigned short*)o0)[(((size_t)bb * NHEAD + hh) * SQ + s) * DHEAD + d] =
              f2bf((f + bias[n]) * scl);
        } else if (n < 2048) {                       // k: raw, [b,h,s,d]
          const int n2 = n - 1024, hh = n2 >> 6, d = n2 & 63;
          ((unsigned short*)o1)[(((size_t)bb * NHEAD + hh) * SQ + s) * DHEAD + d] = f2bf(f);
        } else {                                     // v: +bias, TRANSPOSED [b,h,d,s]
          const int n2 = n - 2048, hh = n2 >> 6, d = n2 & 63;
          ((unsigned short*)o2)[(((size_t)bb * NHEAD + hh) * DHEAD + d) * SQ + s] =
              f2bf(f + bias2[n2]);
        }
      } else if (mode == 1) {                        // pos tables [h, p, d]
        const int hh = n >> 6, d = n & 63;
        const float v = bias ? (f + bias[n]) * scl : f;
        ((unsigned short*)o0)[((size_t)hh * SQ + m) * DHEAD + d] = f2bf(v);
      } else {                                       // out dense + bias + residual (f32)
        ((float*)o0)[(size_t)m * N + n] = f + bias[n] + resid[(size_t)m * N + n];
      }
    }
  }
}

// ---------------------------------------------------------------------------
// Fused disentangled attention: one (b, h, 16-row q-tile) per 128-thread block.
// score[q,k] = q.k + q.PK[clip(q-k+512)] + k.PQ[clip(q-k+512)]  (all via WMMA)
// then masked softmax and ctx = probs @ V (WMMA, V pre-transposed).
// ---------------------------------------------------------------------------
__global__ __launch_bounds__(128) void attn_k(
    const unsigned short* __restrict__ q_s, const unsigned short* __restrict__ k_s,
    const unsigned short* __restrict__ v_t, const unsigned short* __restrict__ posk,
    const unsigned short* __restrict__ posq, const int* __restrict__ amask,
    unsigned short* __restrict__ ctx_bf)
{
  __shared__ float          sc[16][1028];      // score row block (padded)
  __shared__ unsigned short pb16[16][1032];    // probs in bf16 (padded)
  __shared__ unsigned short qs_l[16][72];      // q tile bf16
  __shared__ float          bandC[4][16][33];  // per-wave c2p band (16x32, padded)
  __shared__ float          bandP[4][16][33];  // per-wave p2c band
  __shared__ float          red[16][8];
  __shared__ float          rowstat[16];

  const int tid = threadIdx.x, lane = tid & 31, wave = tid >> 5;
  const int half = lane >> 4, nl = lane & 15;
  const int bh = blockIdx.y, b = bh >> 4, h = bh & 15;
  const int q0 = blockIdx.x * 16;

  { // stage 16x64 q tile into LDS
    const int rr = tid >> 3, cc = (tid & 7) * 8;
    const unsigned short* src = q_s + ((size_t)bh * SQ + q0 + rr) * DHEAD + cc;
#pragma unroll
    for (int i = 0; i < 8; i += 2)
      *(unsigned int*)&qs_l[rr][cc + i] = *(const unsigned int*)(src + i);
  }
  __syncthreads();

  Frag qa0, qa1;                               // q as WMMA A, K split 0..31 / 32..63
  load_frag(&qs_l[nl][0],  lane, qa0);
  load_frag(&qs_l[nl][32], lane, qa1);

  const unsigned short* kb  = k_s  + (size_t)bh * SQ * DHEAD;
  const unsigned short* pkh = posk + (size_t)h  * SQ * DHEAD;
  const unsigned short* pqh = posq + (size_t)h  * SQ * DHEAD;

  for (int it = 0; it < 16; ++it) {            // each wave: interleaved k-tiles
    const int k0 = (it * 4 + wave) * 16;
    const unsigned short* krow = kb + (size_t)(k0 + nl) * DHEAD;
    Frag kf0, kf1;                             // k tile: B frag for qk, A frag for p2c
    load_frag(krow,      lane, kf0);
    load_frag(krow + 32, lane, kf1);

    Acc qk; qk.v = vzero();
    wmma_bf(qk.v, qa0, kf0);
    wmma_bf(qk.v, qa1, kf1);

    const int p0 = q0 - k0 + 512 - 15;         // band base delta
    Acc cb[2], pbv[2];
#pragma unroll
    for (int th = 0; th < 2; ++th) {
      int prow = p0 + th * 16 + nl;
      prow = prow < 0 ? 0 : (prow > 1023 ? 1023 : prow);   // clip() at load
      const unsigned short* pkrow = pkh + (size_t)prow * DHEAD;
      Frag b0, b1;
      load_frag(pkrow,      lane, b0);
      load_frag(pkrow + 32, lane, b1);
      cb[th].v = vzero();
      wmma_bf(cb[th].v, qa0, b0);              // Cb[i][t] = q_i . PK[p0+t]
      wmma_bf(cb[th].v, qa1, b1);
      const unsigned short* pqrow = pqh + (size_t)prow * DHEAD;
      Frag d0, d1;
      load_frag(pqrow,      lane, d0);
      load_frag(pqrow + 32, lane, d1);
      pbv[th].v = vzero();
      wmma_bf(pbv[th].v, kf0, d0);             // Pb[j][t] = k_j . PQ[p0+t]
      wmma_bf(pbv[th].v, kf1, d1);
    }
#pragma unroll
    for (int g = 0; g < 8; ++g) {              // dump bands to per-wave LDS
      const int m = g + 8 * half;
      bandC[wave][m][nl]      = cb[0].f[g];
      bandC[wave][m][16 + nl] = cb[1].f[g];
      bandP[wave][m][nl]      = pbv[0].f[g];
      bandP[wave][m][16 + nl] = pbv[1].f[g];
    }
#pragma unroll
    for (int g = 0; g < 8; ++g) {              // diagonal gather: t = 15 + i - j
      const int m = g + 8 * half;
      const int t = 15 + m - nl;
      sc[m][k0 + nl] = qk.f[g] + bandC[wave][m][t] + bandP[wave][nl][t];
    }
  }
  __syncthreads();

  { // masked softmax: 8 threads per row, stride-8 access
    const int rr = tid >> 3, j0 = tid & 7;
    const int* mrow = amask + ((size_t)b * SQ + q0 + rr) * SQ;
    float mx = -3.4e38f;
    for (int i = 0; i < 128; ++i) {
      const int j = j0 + 8 * i;
      if (mrow[j]) mx = fmaxf(mx, sc[rr][j]);
    }
    red[rr][j0] = mx;
    __syncthreads();
    if (j0 == 0) {
      float m2 = red[rr][0];
#pragma unroll
      for (int t = 1; t < 8; ++t) m2 = fmaxf(m2, red[rr][t]);
      rowstat[rr] = m2;
    }
    __syncthreads();
    const float rmax = rowstat[rr];
    float sum = 0.f;
    for (int i = 0; i < 128; ++i) {
      const int j = j0 + 8 * i;
      const float e = mrow[j] ? __expf(sc[rr][j] - rmax) : 0.f;
      sc[rr][j] = e;
      sum += e;
    }
    red[rr][j0] = sum;
    __syncthreads();
    if (j0 == 0) {
      float s2 = 0.f;
#pragma unroll
      for (int t = 0; t < 8; ++t) s2 += red[rr][t];
      rowstat[rr] = (s2 > 0.f) ? 1.f / s2 : 0.f;
    }
    __syncthreads();
    const float inv = rowstat[rr];
    for (int i = 0; i < 128; ++i) {
      const int j = j0 + 8 * i;
      pb16[rr][j] = f2bf(sc[rr][j] * inv);
    }
  }
  __syncthreads();

  // ctx(16x64) = probs(16x1024) @ V ; wave owns 16 d-columns, V is [b,h,d,s]
  Acc cacc; cacc.v = vzero();
  const unsigned short* vrow = v_t + ((size_t)bh * DHEAD + wave * 16 + nl) * SQ;
  for (int kk = 0; kk < SQ; kk += 32) {
    Frag pa, vf;
    load_frag(&pb16[nl][kk], lane, pa);
    load_frag(vrow + kk,     lane, vf);
    wmma_bf(cacc.v, pa, vf);
  }
#pragma unroll
  for (int g = 0; g < 8; ++g) {
    const int m = g + 8 * half;
    const int d = wave * 16 + nl;
    ctx_bf[((size_t)b * SQ + q0 + m) * 1024 + h * DHEAD + d] = f2bf(cacc.f[g]);
  }
}

// ---------------------------------------------------------------------------
// TF-style LayerNorm, one row (1024) per 128-thread block.
// ---------------------------------------------------------------------------
__global__ __launch_bounds__(128) void ln_k(const float* __restrict__ y,
                                            const float* __restrict__ w,
                                            const float* __restrict__ bb,
                                            float* __restrict__ out)
{
  __shared__ float red[128];
  const int row = blockIdx.x, t = threadIdx.x;
  const float* x = y + (size_t)row * 1024;
  float s = 0.f;
  for (int i = t; i < 1024; i += 128) s += x[i];
  red[t] = s; __syncthreads();
  for (int st = 64; st > 0; st >>= 1) { if (t < st) red[t] += red[t + st]; __syncthreads(); }
  const float mean = red[0] * (1.f / 1024.f);
  __syncthreads();
  float v = 0.f;
  for (int i = t; i < 1024; i += 128) { const float d = x[i] - mean; v += d * d; }
  red[t] = v; __syncthreads();
  for (int st = 64; st > 0; st >>= 1) { if (t < st) red[t] += red[t + st]; __syncthreads(); }
  const float inv = rsqrtf(red[0] * (1.f / 1024.f) + 1e-12f);
  for (int i = t; i < 1024; i += 128)
    out[(size_t)row * 1024 + i] = w[i] * ((x[i] - mean) * inv) + bb[i];
}

extern "C" void kernel_launch(void* const* d_in, const int* in_sizes, int n_in,
                              void* d_out, int out_size, void* d_ws, size_t ws_size,
                              hipStream_t stream) {
  const float* hidden = (const float*)d_in[0];
  const int*   amask  = (const int*)d_in[1];
  const float* relemb = (const float*)d_in[2];
  const float* inw    = (const float*)d_in[3];
  const float* qb     = (const float*)d_in[4];
  const float* vb     = (const float*)d_in[5];
  const float* posw   = (const float*)d_in[6];
  const float* posqw  = (const float*)d_in[7];
  const float* posqb  = (const float*)d_in[8];
  const float* outw   = (const float*)d_in[9];
  const float* outb   = (const float*)d_in[10];
  const float* lnw    = (const float*)d_in[11];
  const float* lnb    = (const float*)d_in[12];

  char* ws = (char*)d_ws;                         // 52 MB total
  unsigned short* q_s  = (unsigned short*)(ws);                       // 8 MB
  unsigned short* k_s  = (unsigned short*)(ws + ((size_t)8  << 20));  // 8 MB
  unsigned short* v_t  = (unsigned short*)(ws + ((size_t)16 << 20));  // 8 MB
  unsigned short* posk = (unsigned short*)(ws + ((size_t)24 << 20));  // 2 MB
  unsigned short* posq = (unsigned short*)(ws + ((size_t)26 << 20));  // 2 MB
  unsigned short* ctxb = (unsigned short*)(ws + ((size_t)28 << 20));  // 8 MB
  float*          ybuf = (float*)         (ws + ((size_t)36 << 20));  // 16 MB

  const float inv_scale = 0.072168784f;           // 1/sqrt(64*3)

  // 1) QKV projection: [4096,1024] @ [3072,1024]^T
  gemm_bf16_k<<<dim3(48, 64), 128, 0, stream>>>(
      hidden, 0, inw, 4096, 3072, 1024, 0, qb, vb, nullptr, inv_scale,
      q_s, k_s, v_t);
  // 2) pos_key = rel_emb @ pos_proj_w^T
  gemm_bf16_k<<<dim3(16, 16), 128, 0, stream>>>(
      relemb, 0, posw, 1024, 1024, 1024, 1, nullptr, nullptr, nullptr, 1.0f,
      posk, nullptr, nullptr);
  // 3) pos_q = (rel_emb @ pos_q_proj_w^T + b) / scale
  gemm_bf16_k<<<dim3(16, 16), 128, 0, stream>>>(
      relemb, 0, posqw, 1024, 1024, 1024, 1, posqb, nullptr, nullptr, inv_scale,
      posq, nullptr, nullptr);
  // 4) fused disentangled attention
  attn_k<<<dim3(64, 64), 128, 0, stream>>>(q_s, k_s, v_t, posk, posq, amask, ctxb);
  // 5) output dense + bias + residual
  gemm_bf16_k<<<dim3(16, 64), 128, 0, stream>>>(
      ctxb, 1, outw, 4096, 1024, 1024, 2, outb, nullptr, hidden, 1.0f,
      ybuf, nullptr, nullptr);
  // 6) LayerNorm
  ln_k<<<4096, 128, 0, stream>>>(ybuf, lnw, lnb, (float*)d_out);
}
